// ResBlockNoConditioning_69999376990851
// MI455X (gfx1250) — matrix operand order repaired and checked by reading, not compiled, not run on hardware
//
#include <hip/hip_runtime.h>
#include <hip/hip_bf16.h>

// ---------------------------------------------------------------------------
// ResBlockNoConditioning fused pipeline for gfx1250 (MI455X, wave32, WMMA)
//   R = 8*4096 = 32768 rows, C = M = K = N = 2048
// ---------------------------------------------------------------------------

typedef __attribute__((ext_vector_type(16))) __bf16 v16bf;
typedef __attribute__((ext_vector_type(8)))  float  v8f;

#define RROWS   32768
#define KDIM    2048
#define NDIM    2048
#define BM      128
#define BN      128
#define BK      64
#define BKPAD   72          // 64 + 8 bf16 pad -> 144B row stride
#define KTILES  (KDIM / BK) // 32

#if (__has_builtin(__builtin_amdgcn_global_load_async_to_lds_b128) && \
     __has_builtin(__builtin_amdgcn_s_wait_asynccnt))
#define HAS_ASYNC_LDS 1
#else
#define HAS_ASYNC_LDS 0
#endif

// exact parameter types of the async-LDS builtin (from the compiler diagnostic):
//   arg0: int __vector_size__(16) __device__ *   (AS1, non-const)
//   arg1: LDS pointer (AS3)
typedef int v4i __attribute__((vector_size(4 * sizeof(int))));
typedef __attribute__((address_space(1))) v4i* v4i_gptr;
typedef __attribute__((address_space(3))) v4i* v4i_lptr;

union Frag16 { uint4 q[2]; v16bf v; };

__device__ __forceinline__ unsigned short f2bf(float f) {
    unsigned u = __builtin_bit_cast(unsigned, f);
    unsigned r = (u + 0x7FFFu + ((u >> 16) & 1u)) >> 16;   // RNE
    return (unsigned short)r;
}

// ---------------------------------------------------------------------------
// Kernel 1: W [k][n] f32  ->  WT [n][k] bf16   (one-time, 16 MB total)
// ---------------------------------------------------------------------------
__global__ __launch_bounds__(256) void k_transpose_bf16(
    const float* __restrict__ W, unsigned short* __restrict__ WT)
{
    int idx = blockIdx.x * 256 + threadIdx.x;     // 2048*2048 threads
    int n = idx >> 11;
    int k = idx & 2047;
    WT[idx] = f2bf(W[(size_t)k * NDIM + n]);
}

// ---------------------------------------------------------------------------
// Kernel 2: wave-per-row LayerNorm + SiLU, f32 -> bf16
// ---------------------------------------------------------------------------
__global__ __launch_bounds__(256) void k_ln_silu_bf16(
    const float* __restrict__ X, const float* __restrict__ g,
    const float* __restrict__ be, unsigned short* __restrict__ Y)
{
    const int wid  = threadIdx.x >> 5;
    const int lane = threadIdx.x & 31;
    const size_t row = (size_t)blockIdx.x * 8 + wid;
    const float* xr = X + row * KDIM;

    float s = 0.f, s2 = 0.f;
    for (int j = lane * 4; j < KDIM; j += 128) {
        float4 v = *(const float4*)(xr + j);
        s  += v.x + v.y + v.z + v.w;
        s2 += v.x * v.x + v.y * v.y + v.z * v.z + v.w * v.w;
    }
    #pragma unroll
    for (int m = 16; m > 0; m >>= 1) {
        s  += __shfl_xor(s,  m, 32);
        s2 += __shfl_xor(s2, m, 32);
    }
    const float mu  = s  * (1.f / KDIM);
    const float var = s2 * (1.f / KDIM) - mu * mu;
    const float rs  = rsqrtf(var + 1e-6f);

    unsigned short* yr = Y + row * KDIM;
    for (int j = lane * 4; j < KDIM; j += 128) {
        float4 v  = *(const float4*)(xr + j);
        float4 gv = *(const float4*)(g  + j);
        float4 bv = *(const float4*)(be + j);
        float h0 = (v.x - mu) * rs * gv.x + bv.x;
        float h1 = (v.y - mu) * rs * gv.y + bv.y;
        float h2 = (v.z - mu) * rs * gv.z + bv.z;
        float h3 = (v.w - mu) * rs * gv.w + bv.w;
        h0 = h0 / (1.f + __expf(-h0));
        h1 = h1 / (1.f + __expf(-h1));
        h2 = h2 / (1.f + __expf(-h2));
        h3 = h3 / (1.f + __expf(-h3));
        ushort4 o; o.x = f2bf(h0); o.y = f2bf(h1); o.z = f2bf(h2); o.w = f2bf(h3);
        *(ushort4*)(yr + j) = o;
    }
}

// ---------------------------------------------------------------------------
// Kernel 3/5: bf16 WMMA GEMM  OUT[r][n] = sum_k A[r][k]*WT[n][k] + bias[n]
//             (+ optional residual).  128x128 block, BK=64, 8 waves.
//             Global->LDS via GLOBAL_LOAD_ASYNC_TO_LDS_B128 (ASYNCcnt),
//             double-buffered LDS; fallback: reg-staged pipeline.
// ---------------------------------------------------------------------------
template <bool RESID>
__global__ __launch_bounds__(256, 1) void k_gemm_bf16(
    const unsigned short* __restrict__ A,    // [RROWS][KDIM] bf16
    const unsigned short* __restrict__ WT,   // [NDIM][KDIM]  bf16
    const float* __restrict__ bias,          // [NDIM]
    const float* __restrict__ resid,         // [RROWS][NDIM] or null
    float* __restrict__ OUT)                 // [RROWS][NDIM]
{
    __shared__ __attribute__((aligned(16))) unsigned short sA[2][BM * BKPAD];
    __shared__ __attribute__((aligned(16))) unsigned short sB[2][BN * BKPAD];

    const int tid   = threadIdx.x;
    const int wid   = tid >> 5;
    const int lane  = tid & 31;
    const int waveM = wid & 3;        // 0..3  -> 32-row slab
    const int waveN = wid >> 2;       // 0..1  -> 64-col slab
    const int lrow  = lane & 15;
    const int lkh   = lane >> 4;      // K-half select

    const int blockN = blockIdx.x * BN;
    const int blockM = blockIdx.y * BM;

    // global->LDS copy plan: each thread owns 4 chunks of 8 bf16 per matrix
    const int srow = tid >> 3;            // 0..31
    const int skc  = (tid & 7) * 8;       // 0..56
    const unsigned short* Ag = A  + (size_t)(blockM + srow) * KDIM + skc;
    const unsigned short* Bg = WT + (size_t)(blockN + srow) * KDIM + skc;

    const v8f vzero = {0.f, 0.f, 0.f, 0.f, 0.f, 0.f, 0.f, 0.f};
    v8f acc[2][4];
    #pragma unroll
    for (int tm = 0; tm < 2; ++tm)
        #pragma unroll
        for (int tn = 0; tn < 4; ++tn) acc[tm][tn] = vzero;

    auto compute = [&](int buf) {
        #pragma unroll
        for (int kk = 0; kk < BK; kk += 32) {
            v16bf aF[2], bF[4];
            #pragma unroll
            for (int tm = 0; tm < 2; ++tm) {
                const int row = waveM * 32 + tm * 16 + lrow;
                const unsigned short* p = &sA[buf][row * BKPAD + kk + lkh * 8];
                Frag16 f;
                f.q[0] = *(const uint4*)(p);        // K = kh*8 .. kh*8+7
                f.q[1] = *(const uint4*)(p + 16);   // K = 16+kh*8 ..
                aF[tm] = f.v;
            }
            #pragma unroll
            for (int tn = 0; tn < 4; ++tn) {
                const int col = waveN * 64 + tn * 16 + lrow;
                const unsigned short* p = &sB[buf][col * BKPAD + kk + lkh * 8];
                Frag16 f;
                f.q[0] = *(const uint4*)(p);        // K = kh*8 .. kh*8+7
                f.q[1] = *(const uint4*)(p + 16);   // K = 16+kh*8 ..
                bF[tn] = f.v;
            }
            #pragma unroll
            for (int tm = 0; tm < 2; ++tm)
                #pragma unroll
                for (int tn = 0; tn < 4; ++tn)
                    acc[tm][tn] = __builtin_amdgcn_wmma_f32_16x16x32_bf16(
                        false, aF[tm], false, bF[tn],
                        (short)0, acc[tm][tn], false, false);
        }
    };

#if HAS_ASYNC_LDS
    // ---------------- async global->LDS path (ASYNCcnt) ----------------
    auto issue_tile = [&](int kt, int buf) {
        #pragma unroll
        for (int i = 0; i < 4; ++i) {
            __builtin_amdgcn_global_load_async_to_lds_b128(
                (v4i_gptr)(Ag + (size_t)i * 32 * KDIM + kt),
                (v4i_lptr)&sA[buf][(srow + i * 32) * BKPAD + skc],
                0, 0);
            __builtin_amdgcn_global_load_async_to_lds_b128(
                (v4i_gptr)(Bg + (size_t)i * 32 * KDIM + kt),
                (v4i_lptr)&sB[buf][(srow + i * 32) * BKPAD + skc],
                0, 0);
        }
    };

    issue_tile(0, 0);
    for (int t = 0; t < KTILES; ++t) {
        if (t + 1 < KTILES) {
            issue_tile((t + 1) * BK, (t + 1) & 1);
            if (t + 2 < KTILES) {
                __builtin_prefetch(Ag + (t + 2) * BK, 0, 1);
                __builtin_prefetch(Bg + (t + 2) * BK, 0, 1);
            }
            __builtin_amdgcn_s_wait_asynccnt(8);   // tile t's 8 copies done
        } else {
            __builtin_amdgcn_s_wait_asynccnt(0);
        }
        __syncthreads();                           // all waves' tile t in LDS
        compute(t & 1);
        __syncthreads();                           // buf free before t+2 lands
    }
#else
    // ---------------- fallback: reg-staged pipeline ----------------
    uint4 ra[4], rb[4];
    auto gload = [&](int kt) {
        #pragma unroll
        for (int i = 0; i < 4; ++i) {
            ra[i] = *(const uint4*)(Ag + (size_t)i * 32 * KDIM + kt);
            rb[i] = *(const uint4*)(Bg + (size_t)i * 32 * KDIM + kt);
        }
    };
    auto sstore = [&](int buf) {
        #pragma unroll
        for (int i = 0; i < 4; ++i) {
            *(uint4*)&sA[buf][(srow + i * 32) * BKPAD + skc] = ra[i];
            *(uint4*)&sB[buf][(srow + i * 32) * BKPAD + skc] = rb[i];
        }
    };

    gload(0);
    sstore(0);
    __syncthreads();
    for (int t = 0; t < KTILES; ++t) {
        if (t + 1 < KTILES) gload((t + 1) * BK);
        if (t + 2 < KTILES) {
            __builtin_prefetch(Ag + (t + 2) * BK, 0, 1);
            __builtin_prefetch(Bg + (t + 2) * BK, 0, 1);
        }
        compute(t & 1);
        if (t + 1 < KTILES) sstore((t + 1) & 1);
        __syncthreads();
    }
#endif

    // epilogue: lane holds column n = lrow of each 16x16 tile, rows lkh*8 .. +7
    #pragma unroll
    for (int tm = 0; tm < 2; ++tm) {
        const int rowbase = blockM + waveM * 32 + tm * 16 + lkh * 8;
        #pragma unroll
        for (int tn = 0; tn < 4; ++tn) {
            const int col = blockN + waveN * 64 + tn * 16 + lrow;
            const float bv = bias[col];
            float* op = OUT + (size_t)rowbase * NDIM + col;
            #pragma unroll
            for (int i = 0; i < 8; ++i) {
                float v = acc[tm][tn][i] + bv;
                if constexpr (RESID) v += resid[(size_t)(rowbase + i) * NDIM + col];
                op[(size_t)i * NDIM] = v;
            }
        }
    }
}

// ---------------------------------------------------------------------------
extern "C" void kernel_launch(void* const* d_in, const int* in_sizes, int n_in,
                              void* d_out, int out_size, void* d_ws, size_t ws_size,
                              hipStream_t stream)
{
    (void)in_sizes; (void)n_in; (void)out_size; (void)ws_size;
    const float* x   = (const float*)d_in[0];
    const float* w1  = (const float*)d_in[1];
    const float* b1  = (const float*)d_in[2];
    const float* w2  = (const float*)d_in[3];
    const float* b2  = (const float*)d_in[4];
    const float* g1  = (const float*)d_in[5];
    const float* be1 = (const float*)d_in[6];
    const float* g2  = (const float*)d_in[7];
    const float* be2 = (const float*)d_in[8];
    float* out = (float*)d_out;

    char* ws = (char*)d_ws;
    unsigned short* WT1 = (unsigned short*)(ws);                    //  8 MB
    unsigned short* WT2 = (unsigned short*)(ws + 8388608);          //  8 MB
    unsigned short* Abf = (unsigned short*)(ws + 16777216);         // 128 MB

    // 1) weights -> bf16, pre-transposed [n][k]
    k_transpose_bf16<<<16384, 256, 0, stream>>>(w1, WT1);
    k_transpose_bf16<<<16384, 256, 0, stream>>>(w2, WT2);

    // 2) LN1 + SiLU -> bf16 activations
    k_ln_silu_bf16<<<RROWS / 8, 256, 0, stream>>>(x, g1, be1, Abf);

    // 3) GEMM1 (+b1) -> d_out holds fp32 H
    k_gemm_bf16<false><<<dim3(NDIM / BN, RROWS / BM), 256, 0, stream>>>(
        Abf, WT1, b1, nullptr, out);

    // 4) LN2 + SiLU on H -> bf16 (reuse Abf)
    k_ln_silu_bf16<<<RROWS / 8, 256, 0, stream>>>(out, g2, be2, Abf);

    // 5) GEMM2 (+b2) + residual x -> final d_out
    k_gemm_bf16<true><<<dim3(NDIM / BN, RROWS / BM), 256, 0, stream>>>(
        Abf, WT2, b2, x, out);
}
